// DeformConv_2869038154543
// MI455X (gfx1250) — matrix operand order repaired and checked by reading, not compiled, not run on hardware
//
#include <hip/hip_runtime.h>

// ---------------- problem constants ----------------
#define B_    2
#define CHI_  256
#define CHO_  256
#define H_    128
#define W_    128
#define HW_   (H_*W_)
#define NPIX  (B_*H_*W_)       // 32768
#define BN_EPS 1e-5f

// ---------------- vector types ----------------
typedef __attribute__((ext_vector_type(8)))  unsigned short ushort8;
typedef __attribute__((ext_vector_type(16))) unsigned short ushort16;
typedef __attribute__((ext_vector_type(16))) __bf16         bf16x16;
typedef __attribute__((ext_vector_type(8)))  float          v8f;

__device__ __forceinline__ unsigned short f2bf(float f) {
  unsigned int u = __builtin_bit_cast(unsigned int, f);
  u += 0x7fffu + ((u >> 16) & 1u);          // round-to-nearest-even
  return (unsigned short)(u >> 16);
}
__device__ __forceinline__ float bf2f(unsigned short h) {
  unsigned int u = ((unsigned int)h) << 16;
  return __builtin_bit_cast(float, u);
}
__device__ __forceinline__ bf16x16 pack16(ushort8 a, ushort8 b) {
  ushort16 u = __builtin_shufflevector(a, b, 0,1,2,3,4,5,6,7,8,9,10,11,12,13,14,15);
  return __builtin_bit_cast(bf16x16, u);
}

// =====================================================================
// Prep kernel: x NCHW f32 -> NHWC bf16  (corner gathers become contiguous)
// =====================================================================
__global__ void __launch_bounds__(256)
k_xpack(const float* __restrict__ x, unsigned short* __restrict__ xbf) {
  int idx = blockIdx.x * 256 + threadIdx.x;       // 8,388,608 total
  int c  =  idx        & 255;
  int xx = (idx >> 8)  & 127;
  int yy = (idx >> 15) & 127;
  int b  =  idx >> 22;
  xbf[idx] = f2bf(x[((b*CHI_ + c)*H_ + yy)*W_ + xx]);
}

// =====================================================================
// Prep: pack w_dcn into per-lane WMMA A-fragment order (bf16)
// layout: [t(9)][cc(8)][otile(16)][lane(32)][e(16)]
// lane<16: K = K0+e (e<8) / K0+16+(e-8),  K0 = 0 ;  lane>=16: K0 = 8
// =====================================================================
__global__ void __launch_bounds__(256)
k_wpack2(const float* __restrict__ w, unsigned short* __restrict__ wf) {
  int idx  = blockIdx.x * 256 + threadIdx.x;      // 589,824 total
  int e    =  idx        & 15;
  int lane = (idx >> 4)  & 31;
  int ot   = (idx >> 9)  & 15;
  int cc   = (idx >> 13) & 7;
  int t    =  idx >> 16;
  int K0   = (lane < 16) ? 0 : 8;
  int kl   = (e < 8) ? (K0 + e) : (K0 + 16 + (e - 8));
  int c    = cc*32 + kl;
  int o    = ot*16 + (lane & 15);
  wf[idx] = f2bf(w[((o*CHI_ + c)*3 + t/3)*3 + (t%3)]);
}

// same for w_off, padded 27 -> 32 output rows: [t(9)][cc(8)][otile(2)][lane][e]
__global__ void __launch_bounds__(256)
k_wpack1(const float* __restrict__ w, unsigned short* __restrict__ wf) {
  int idx  = blockIdx.x * 256 + threadIdx.x;      // 73,728 total
  int e    =  idx        & 15;
  int lane = (idx >> 4)  & 31;
  int ot   = (idx >> 9)  & 1;
  int cc   = (idx >> 10) & 7;
  int t    =  idx >> 13;
  int K0   = (lane < 16) ? 0 : 8;
  int kl   = (e < 8) ? (K0 + e) : (K0 + 16 + (e - 8));
  int c    = cc*32 + kl;
  int o    = ot*16 + (lane & 15);
  float v  = (o < 27) ? w[((o*CHI_ + c)*3 + t/3)*3 + (t%3)] : 0.0f;
  wf[idx] = f2bf(v);
}

// fold conv-bias + BN into per-channel scale/bias
__global__ void k_bnprep(const float* __restrict__ b_dcn, const float* __restrict__ gamma,
                         const float* __restrict__ beta,  const float* __restrict__ mean,
                         const float* __restrict__ var,
                         float* __restrict__ scale, float* __restrict__ bias) {
  int o = threadIdx.x;
  float s = gamma[o] * rsqrtf(var[o] + BN_EPS);
  scale[o] = s;
  bias[o]  = (b_dcn[o] - mean[o]) * s + beta[o];
}

// =====================================================================
// Offset conv (256 -> 27ch, padded 32) as WMMA implicit GEMM.
// im2col copy uses CDNA5 async global->LDS DMA (ASYNCcnt).
// Block: 256 thr = 8 waves, tile M=32 x N=64 pixels, K=2304 (9 taps x 8 cchunks)
// =====================================================================
__global__ void __launch_bounds__(256)
k_offconv(const unsigned short* __restrict__ xbf,
          const unsigned short* __restrict__ wfrag,
          const float* __restrict__ b_off,
          float* __restrict__ om) {
  __shared__ unsigned short col[64 * 264];        // [pix][c], stride 264 avoids bank conflicts

  const int tid  = threadIdx.x;
  const int lane = tid & 31;
  const int wid  = tid >> 5;
  const int nt   = wid & 3;                       // 4 pixel sub-tiles of 16
  const int ot   = wid >> 2;                      // 2 output sub-tiles of 16
  const int p0   = blockIdx.x * 64;

  v8f acc = {0.f,0.f,0.f,0.f,0.f,0.f,0.f,0.f};

  const int ci  = tid >> 2;                       // im2col: pixel 0..63
  const int cst = (tid & 3) * 64;                 // channel start
  const int gp  = p0 + ci;
  const int px  = gp & 127, py = (gp >> 7) & 127, pb = gp >> 14;

  const int kb  = (lane < 16) ? 0 : 8;
  const int row = nt*16 + (lane & 15);

  for (int t = 0; t < 9; ++t) {
    __syncthreads();
    int yy = py + t/3 - 1;
    int xx = px + t%3 - 1;
    bool ok = (yy >= 0) && (yy < H_) && (xx >= 0) && (xx < W_);
    if (ok) {
      const unsigned short* src = xbf + (((pb*H_ + yy)*W_ + xx) << 8);
      #pragma unroll
      for (int q = 0; q < 8; ++q) {
        int c = cst + q*8;
        unsigned int       ldso = (unsigned int)(unsigned long long)&col[ci*264 + c];
        unsigned long long ga   = (unsigned long long)(src + c);
        // CDNA5 async DMA: 16B global -> LDS, tracked by ASYNCcnt
        asm volatile("global_load_async_to_lds_b128 %0, %1, off"
                     :: "v"(ldso), "v"(ga) : "memory");
      }
    } else {
      uint4 z = {0u, 0u, 0u, 0u};
      #pragma unroll
      for (int q = 0; q < 8; ++q)
        *(uint4*)&col[ci*264 + cst + q*8] = z;
    }
    asm volatile("s_wait_asynccnt 0" ::: "memory");
    __syncthreads();
    #pragma unroll
    for (int cc = 0; cc < 8; ++cc) {
      const unsigned short* wp = wfrag + ((((t*8 + cc)*2 + ot) << 9) + lane*16);
      bf16x16 am = pack16(*(const ushort8*)wp, *(const ushort8*)(wp + 8));
      ushort8 b0 = *(const ushort8*)&col[row*264 + cc*32 + kb];
      ushort8 b1 = *(const ushort8*)&col[row*264 + cc*32 + kb + 16];
      bf16x16 bm = pack16(b0, b1);
      acc = __builtin_amdgcn_wmma_f32_16x16x32_bf16(false, am, false, bm,
                                                    (short)0, acc, false, false);
    }
  }

  // epilogue: write raw om[b][27][H][W]
  const int g  = p0 + nt*16 + (lane & 15);
  const int gx = g & 127, gy = (g >> 7) & 127, gb = g >> 14;
  #pragma unroll
  for (int r = 0; r < 8; ++r) {
    int o = ot*16 + r + ((lane < 16) ? 0 : 8);
    if (o < 27)
      om[((gb*27 + o)*H_ + gy)*W_ + gx] = acc[r] + b_off[o];
  }
}

// =====================================================================
// Main DCNv2 GEMM with on-the-fly bilinear sampling.
// Grid: (512 pixel tiles of 64, 2 blocks of 128 output channels)
// Block 256 thr = 8 waves: 4 M-groups (32 ch) x 2 N-groups (32 pix).
// Each wave: 2 M-subtiles x 2 N-subtiles = 4 accumulators, A-frags reused
// across both N-subtiles (halves weight traffic vs 32-pixel tiles).
// =====================================================================
__global__ void __launch_bounds__(256)
k_dcn(const unsigned short* __restrict__ xbf,
      const unsigned short* __restrict__ wfrag,
      const float* __restrict__ om,
      const float* __restrict__ scale,
      const float* __restrict__ bias,
      float* __restrict__ out) {
  __shared__ unsigned short samp[64 * 264];       // sampled B tile [pix][c]  (33.8 KB)
  __shared__ int   cidx[9][64][4];                // corner base (elem index, c=0)
  __shared__ float cwt [9][64][4];                // bilinear*valid*mask weight

  const int tid  = threadIdx.x;
  const int lane = tid & 31;
  const int wid  = tid >> 5;
  const int wm   = wid & 3;                       // M group (32 channels)
  const int wn   = wid >> 2;                      // N group (32 pixels)
  const int p0   = blockIdx.x * 64;
  const int obase = blockIdx.y * 8;               // base otile (16 per full M)

  // ---- per-(tap,pixel) bilinear corner precompute ----
  for (int j = tid; j < 576; j += 256) {
    int t = j >> 6, i = j & 63;
    int g = p0 + i;
    int gx = g & 127, gy = (g >> 7) & 127, gb = g >> 14;
    int base = (gb*27)*HW_ + gy*W_ + gx;
    float offy = om[base + (2*t  )*HW_];
    float offx = om[base + (2*t+1)*HW_];
    float m    = om[base + (18+t )*HW_];
    m = 1.0f / (1.0f + __expf(-m));               // sigmoid mask
    float pyf = (float)gy - 1.0f + (float)(t/3) + offy;
    float pxf = (float)gx - 1.0f + (float)(t%3) + offx;
    float y0f = floorf(pyf), x0f = floorf(pxf);
    int   y0  = (int)y0f,    x0  = (int)x0f;
    float wy1 = pyf - y0f, wy0 = 1.f - wy1;
    float wx1 = pxf - x0f, wx0 = 1.f - wx1;
    #pragma unroll
    for (int cnr = 0; cnr < 4; ++cnr) {
      int yi = y0 + (cnr >> 1);
      int xi = x0 + (cnr & 1);
      bool ok = (yi >= 0) && (yi < H_) && (xi >= 0) && (xi < W_);
      float w = ((cnr >> 1) ? wy1 : wy0) * ((cnr & 1) ? wx1 : wx0) * m;
      cidx[t][i][cnr] = ok ? (((gb*H_ + yi)*W_ + xi) << 8) : 0;
      cwt [t][i][cnr] = ok ? w : 0.f;
    }
  }

  v8f acc00 = {0.f,0.f,0.f,0.f,0.f,0.f,0.f,0.f};  // (M0,N0)
  v8f acc01 = {0.f,0.f,0.f,0.f,0.f,0.f,0.f,0.f};  // (M0,N1)
  v8f acc10 = {0.f,0.f,0.f,0.f,0.f,0.f,0.f,0.f};  // (M1,N0)
  v8f acc11 = {0.f,0.f,0.f,0.f,0.f,0.f,0.f,0.f};  // (M1,N1)

  const int si = tid >> 2;                        // sampler: pixel 0..63
  const int sc = (tid & 3) * 64;                  // channel start (64 per thread)
  const int kb   = (lane < 16) ? 0 : 8;
  const int row0 = wn*32 + (lane & 15);
  const int row1 = row0 + 16;

  for (int t = 0; t < 9; ++t) {
    __builtin_prefetch(wfrag + (((t*8*16 + obase) << 9)), 0, 1);
    __syncthreads();
    // ---- build sampled tile: samp[pix][c] = bilinear(x) * mask (bf16) ----
    int   i0 = cidx[t][si][0], i1 = cidx[t][si][1], i2 = cidx[t][si][2], i3 = cidx[t][si][3];
    float w0 = cwt [t][si][0], w1 = cwt [t][si][1], w2 = cwt [t][si][2], w3 = cwt [t][si][3];
    #pragma unroll
    for (int q = 0; q < 32; ++q) {
      int c = sc + q*2;
      unsigned int v0 = *(const unsigned int*)(xbf + i0 + c);
      unsigned int v1 = *(const unsigned int*)(xbf + i1 + c);
      unsigned int v2 = *(const unsigned int*)(xbf + i2 + c);
      unsigned int v3 = *(const unsigned int*)(xbf + i3 + c);
      float lo = w0*bf2f((unsigned short)v0) + w1*bf2f((unsigned short)v1)
               + w2*bf2f((unsigned short)v2) + w3*bf2f((unsigned short)v3);
      float hi = w0*bf2f((unsigned short)(v0>>16)) + w1*bf2f((unsigned short)(v1>>16))
               + w2*bf2f((unsigned short)(v2>>16)) + w3*bf2f((unsigned short)(v3>>16));
      unsigned int pk = (unsigned int)f2bf(lo) | ((unsigned int)f2bf(hi) << 16);
      *(unsigned int*)&samp[si*264 + c] = pk;
    }
    __syncthreads();
    // ---- 8 K-chunks of 32 channels: 4 WMMAs each (2 M-tiles x 2 N-tiles) ----
    #pragma unroll
    for (int cc = 0; cc < 8; ++cc) {
      bf16x16 bm0 = pack16(*(const ushort8*)&samp[row0*264 + cc*32 + kb],
                           *(const ushort8*)&samp[row0*264 + cc*32 + kb + 16]);
      bf16x16 bm1 = pack16(*(const ushort8*)&samp[row1*264 + cc*32 + kb],
                           *(const ushort8*)&samp[row1*264 + cc*32 + kb + 16]);
      const int otg = obase + wm*2;
      const unsigned short* wp = wfrag + ((((t*8 + cc)*16 + otg) << 9) + lane*16);
      bf16x16 a0 = pack16(*(const ushort8*)wp,         *(const ushort8*)(wp + 8));
      bf16x16 a1 = pack16(*(const ushort8*)(wp + 512), *(const ushort8*)(wp + 520));
      acc00 = __builtin_amdgcn_wmma_f32_16x16x32_bf16(false, a0, false, bm0,
                                                      (short)0, acc00, false, false);
      acc01 = __builtin_amdgcn_wmma_f32_16x16x32_bf16(false, a0, false, bm1,
                                                      (short)0, acc01, false, false);
      acc10 = __builtin_amdgcn_wmma_f32_16x16x32_bf16(false, a1, false, bm0,
                                                      (short)0, acc10, false, false);
      acc11 = __builtin_amdgcn_wmma_f32_16x16x32_bf16(false, a1, false, bm1,
                                                      (short)0, acc11, false, false);
    }
  }

  // ---- fused BN + ReLU epilogue, coalesced NCHW stores ----
  const int g0  = p0 + wn*32 + (lane & 15);
  const int g1  = g0 + 16;
  const int gx0 = g0 & 127, gy0 = (g0 >> 7) & 127, gb0 = g0 >> 14;
  const int gx1 = g1 & 127, gy1 = (g1 >> 7) & 127, gb1 = g1 >> 14;
  #pragma unroll
  for (int r = 0; r < 8; ++r) {
    int o0 = blockIdx.y*128 + wm*32 + r + ((lane < 16) ? 0 : 8);
    int o1 = o0 + 16;
    float s0 = scale[o0], c0 = bias[o0];
    float s1 = scale[o1], c1 = bias[o1];
    out[((gb0*CHO_ + o0)*H_ + gy0)*W_ + gx0] = fmaxf(acc00[r]*s0 + c0, 0.f);
    out[((gb1*CHO_ + o0)*H_ + gy1)*W_ + gx1] = fmaxf(acc01[r]*s0 + c0, 0.f);
    out[((gb0*CHO_ + o1)*H_ + gy0)*W_ + gx0] = fmaxf(acc10[r]*s1 + c1, 0.f);
    out[((gb1*CHO_ + o1)*H_ + gy1)*W_ + gx1] = fmaxf(acc11[r]*s1 + c1, 0.f);
  }
}

// =====================================================================
// host launcher
// =====================================================================
extern "C" void kernel_launch(void* const* d_in, const int* in_sizes, int n_in,
                              void* d_out, int out_size, void* d_ws, size_t ws_size,
                              hipStream_t stream) {
  const float* x      = (const float*)d_in[0];
  const float* w_off  = (const float*)d_in[1];
  const float* b_off  = (const float*)d_in[2];
  const float* w_dcn  = (const float*)d_in[3];
  const float* b_dcn  = (const float*)d_in[4];
  const float* gamma  = (const float*)d_in[5];
  const float* beta   = (const float*)d_in[6];
  const float* bnmean = (const float*)d_in[7];
  const float* bnvar  = (const float*)d_in[8];
  float* out = (float*)d_out;

  // workspace carve-up (bytes)
  char* ws = (char*)d_ws;
  unsigned short* xbf = (unsigned short*)(ws);                        // 16,777,216
  unsigned short* wf2 = (unsigned short*)(ws + 16777216);             //  1,179,648
  unsigned short* wf1 = (unsigned short*)(ws + 16777216 + 1179648);   //    147,456
  float* om    = (float*)(ws + 18104320);                             //  3,538,944
  float* scale = (float*)(ws + 21643264);                             //      1,024
  float* biasv = (float*)(ws + 21644288);                             //      1,024

  k_xpack <<<dim3(32768), dim3(256), 0, stream>>>(x, xbf);
  k_wpack2<<<dim3(2304),  dim3(256), 0, stream>>>(w_dcn, wf2);
  k_wpack1<<<dim3(288),   dim3(256), 0, stream>>>(w_off, wf1);
  k_bnprep<<<dim3(1),     dim3(256), 0, stream>>>(b_dcn, gamma, beta, bnmean, bnvar,
                                                  scale, biasv);
  k_offconv<<<dim3(512),    dim3(256), 0, stream>>>(xbf, wf1, b_off, om);
  k_dcn    <<<dim3(512, 2), dim3(256), 0, stream>>>(xbf, wf2, om, scale, biasv, out);
}
